// CapRNNModelHelper_70102456205953
// MI455X (gfx1250) — matrix operand order repaired and checked by reading, not compiled, not run on hardware
//
#include <hip/hip_runtime.h>
#include <hip/hip_bf16.h>
#include <stdint.h>

// Problem constants
#define B_    128
#define S_    256
#define DW_   300
#define DWP_  320      // D_W padded to multiple of 32 (WMMA K chunk)
#define H_    128
#define G3_   384      // 3*H (gates r,z,n)
#define NC_   10
#define DC_   16
#define ROWS_ 32768    // S*B

typedef __attribute__((ext_vector_type(16))) _Float16 v16h;
typedef __attribute__((ext_vector_type(8)))  _Float16 v8h;
typedef __attribute__((ext_vector_type(8)))  float    v8f;

// ---------------------------------------------------------------------------
// Tensor Data Mover: 2-D tile load Global -> LDS (CDNA5 TDM, CDNA5 ISA)
#if __has_builtin(__builtin_amdgcn_tensor_load_to_lds) && __has_builtin(__builtin_amdgcn_s_wait_tensorcnt)
#define HAVE_TDM 1
#else
#define HAVE_TDM 0
#endif

#if HAVE_TDM
typedef unsigned int u32x4 __attribute__((ext_vector_type(4)));
typedef int          i32x4 __attribute__((ext_vector_type(4)));
typedef int          i32x8 __attribute__((ext_vector_type(8)));

// Loads a tile_d1 x tile_d0 tile of 2-byte elements into LDS (rows packed
// contiguously: lds[row*tile_d0 + k]). stride0 = tensor row pitch in elements.
__device__ __forceinline__ void tdm_load_2d(void* lds, const void* gptr,
                                            uint32_t tile_d0, uint32_t tile_d1,
                                            uint32_t stride0) {
  uint64_t ga = (uint64_t)(uintptr_t)gptr;          // global byte address (57b)
  uint32_t lo = (uint32_t)(uintptr_t)lds;           // LDS addr = low 32 bits
  // D# group 0: [1:0]=count=1 | lds_addr | global_addr[56:0] | type=2
  u32x4 g0 = { 1u, lo, (uint32_t)ga,
               (uint32_t)((ga >> 32) & 0x01FFFFFFu) | (2u << 30) };
  // D# group 1 bitfields (ISA 8.4): data_size=1 (2B); tiles always in-bounds,
  // so tensor dims just need to cover the tile.
  uint32_t td0 = stride0;          // tensor_dim0
  uint32_t td1 = 0x00100000u;      // tensor_dim1 (large)
  i32x8 g1;
  g1[0] = (int)(1u << 16);                                             // data_size
  g1[1] = (int)((td0 & 0xFFFFu) << 16);                                // dim0[15:0] @48
  g1[2] = (int)(((td0 >> 16) & 0xFFFFu) | ((td1 & 0xFFFFu) << 16));    // dim0[31:16], dim1[15:0]
  g1[3] = (int)(((td1 >> 16) & 0xFFFFu) | ((tile_d0 & 0xFFFFu) << 16));// dim1[31:16], tile_dim0
  g1[4] = (int)(tile_d1 & 0xFFFFu);                                    // tile_dim1, tile_dim2=0
  g1[5] = (int)stride0;                                                // dim0_stride[31:0]
  g1[6] = 0;                                                           // dim0_stride[47:32], dim1_stride lo
  g1[7] = 0;
  i32x4 z4 = {0, 0, 0, 0};
#if defined(__clang_major__) && (__clang_major__ >= 23)
  i32x8 z8 = {0, 0, 0, 0, 0, 0, 0, 0};
  __builtin_amdgcn_tensor_load_to_lds(g0, g1, z4, z4, z8, 0);
#else
  __builtin_amdgcn_tensor_load_to_lds(g0, g1, z4, z4, 0);
#endif
}
#endif  // HAVE_TDM

// ---------------------------------------------------------------------------
// WMMA fragment helpers (wave32, V_WMMA_F32_16X16X32_F16)
__device__ __forceinline__ v16h frag_a_rm(const _Float16* p, int ld) {
  int lane = threadIdx.x & 31;
  const _Float16* r = p + (lane & 15) * ld + ((lane & 16) ? 8 : 0);
  v8h lo = *(const v8h*)(r);
  v8h hi = *(const v8h*)(r + 16);
  v16h a;
#pragma unroll
  for (int e = 0; e < 8; ++e) { a[e] = lo[e]; a[8 + e] = hi[e]; }
  return a;
}

__device__ __forceinline__ v16h frag_b_nt(const _Float16* p, int ld) {
  int lane = threadIdx.x & 31;
  const _Float16* r = p + (lane & 15) * ld + ((lane & 16) ? 8 : 0);
  v8h lo = *(const v8h*)(r);
  v8h hi = *(const v8h*)(r + 16);
  v16h b;
#pragma unroll
  for (int e = 0; e < 8; ++e) { b[e] = lo[e]; b[8 + e] = hi[e]; }
  return b;
}

__device__ __forceinline__ v8f wmma_f16(v16h a, v16h b, v8f c) {
  return __builtin_amdgcn_wmma_f32_16x16x32_f16(
      false, a, false, b, (short)0, c, false, false);
}

// ---------------------------------------------------------------------------
// K0: convert weights to f16 (w_ih padded 300->320, W_cap transposed)
__global__ void prep_weights(const float* __restrict__ wihF, const float* __restrict__ wihB,
                             const float* __restrict__ whhF, const float* __restrict__ whhB,
                             const float* __restrict__ Wcap,
                             _Float16* __restrict__ wih16, _Float16* __restrict__ whh16,
                             _Float16* __restrict__ wcap16t) {
  int stride = gridDim.x * blockDim.x;
  int gtid = blockIdx.x * blockDim.x + threadIdx.x;
  for (int idx = gtid; idx < 2 * G3_ * DWP_; idx += stride) {
    int dir = idx / (G3_ * DWP_);
    int rem = idx - dir * (G3_ * DWP_);
    int g = rem / DWP_, k = rem - g * DWP_;
    const float* src = dir ? wihB : wihF;
    wih16[idx] = (k < DW_) ? (_Float16)src[g * DW_ + k] : (_Float16)0.f;
  }
  for (int idx = gtid; idx < 2 * G3_ * H_; idx += stride) {
    int dir = idx / (G3_ * H_);
    int rem = idx - dir * (G3_ * H_);
    const float* src = dir ? whhB : whhF;
    whh16[idx] = (_Float16)src[rem];
  }
  for (int idx = gtid; idx < 160 * 256; idx += stride) {
    int m = idx >> 8, d = idx & 255;
    wcap16t[idx] = (_Float16)Wcap[d * 160 + m];  // W_cap is [256][160]
  }
}

// ---------------------------------------------------------------------------
// K1: embedding gather -> f16, layout [s][b][320] (row r = s*B + b), zero pad
__global__ void embed_gather(const int* __restrict__ x, const float* __restrict__ emb,
                             _Float16* __restrict__ ef16) {
  int stride = gridDim.x * blockDim.x;
  for (int idx = blockIdx.x * blockDim.x + threadIdx.x; idx < ROWS_ * DWP_; idx += stride) {
    int r = idx / DWP_, k = idx - r * DWP_;
    int s = r >> 7, b = r & 127;  // B_ == 128
    float v = 0.f;
    if (k < DW_) v = emb[(size_t)x[b * S_ + s] * DW_ + k];
    ef16[idx] = (_Float16)v;
  }
}

// ---------------------------------------------------------------------------
// K2: x_proj GEMM: [32768 x 320] x [384 x 320]^T + b_ih -> f32 [dir][32768][384]
// Block tile 128x64, 8 waves, 2x2 WMMA tiles each. K chunks double-buffered
// in LDS via TDM (fallback: manual staging).
__global__ __launch_bounds__(256) void xproj_gemm(
    const _Float16* __restrict__ ef16, const _Float16* __restrict__ wih16,
    const float* __restrict__ bihF, const float* __restrict__ bihB,
    float* __restrict__ xproj) {
  __shared__ __attribute__((aligned(64))) _Float16 As[2][128 * 32];
  __shared__ __attribute__((aligned(64))) _Float16 Bs[2][64 * 32];

  int tid = threadIdx.x, lane = tid & 31, w = tid >> 5;
  int wm = w & 3, wn = w >> 2;
  int dir = blockIdx.z;
  size_t r0 = (size_t)blockIdx.x * 128;
  int n0g = blockIdx.y * 64;
  const _Float16* A = ef16 + r0 * DWP_;
  const _Float16* Bw = wih16 + (size_t)dir * G3_ * DWP_ + (size_t)n0g * DWP_;
  const float* bih = dir ? bihB : bihF;

  v8f zz = {};
  v8f acc[2][2];
#pragma unroll
  for (int i = 0; i < 2; ++i)
#pragma unroll
    for (int j = 0; j < 2; ++j) acc[i][j] = zz;

#if HAVE_TDM
  if (w == 0) {  // TDM ignores EXEC; issue once from wave 0, track TENSORcnt
    tdm_load_2d(As[0], A, 32, 128, DWP_);
    tdm_load_2d(Bs[0], Bw, 32, 64, DWP_);
    __builtin_amdgcn_s_wait_tensorcnt(0);
  }
  __syncthreads();
  for (int kc = 0; kc < 10; ++kc) {
    int cur = kc & 1, nxt = cur ^ 1;
    if (w == 0 && kc + 1 < 10) {  // DMA next chunk while WMMAs run
      tdm_load_2d(As[nxt], A + (kc + 1) * 32, 32, 128, DWP_);
      tdm_load_2d(Bs[nxt], Bw + (kc + 1) * 32, 32, 64, DWP_);
    }
    v16h a0 = frag_a_rm(As[cur] + (wm * 32 + 0) * 32, 32);
    v16h a1 = frag_a_rm(As[cur] + (wm * 32 + 16) * 32, 32);
    v16h b0 = frag_b_nt(Bs[cur] + (wn * 32 + 0) * 32, 32);
    v16h b1 = frag_b_nt(Bs[cur] + (wn * 32 + 16) * 32, 32);
    acc[0][0] = wmma_f16(a0, b0, acc[0][0]);
    acc[0][1] = wmma_f16(a0, b1, acc[0][1]);
    acc[1][0] = wmma_f16(a1, b0, acc[1][0]);
    acc[1][1] = wmma_f16(a1, b1, acc[1][1]);
    __syncthreads();                       // reads of cur done; safe to overwrite
    if (w == 0 && kc + 1 < 10) __builtin_amdgcn_s_wait_tensorcnt(0);
    __syncthreads();                       // nxt buffer now visible to all waves
  }
#else
  int arow = tid >> 1, aseg = tid & 1;
  for (int kc = 0; kc < 10; ++kc) {
    *(v16h*)(As[0] + arow * 32 + aseg * 16) =
        *(const v16h*)(A + (size_t)arow * DWP_ + kc * 32 + aseg * 16);
    if (tid < 128) {
      *(v16h*)(Bs[0] + arow * 32 + aseg * 16) =
          *(const v16h*)(Bw + (size_t)arow * DWP_ + kc * 32 + aseg * 16);
    }
    if (kc + 1 < 10)
      __builtin_prefetch(A + (size_t)arow * DWP_ + (kc + 1) * 32, 0, 3);
    __syncthreads();
    v16h a0 = frag_a_rm(As[0] + (wm * 32 + 0) * 32, 32);
    v16h a1 = frag_a_rm(As[0] + (wm * 32 + 16) * 32, 32);
    v16h b0 = frag_b_nt(Bs[0] + (wn * 32 + 0) * 32, 32);
    v16h b1 = frag_b_nt(Bs[0] + (wn * 32 + 16) * 32, 32);
    acc[0][0] = wmma_f16(a0, b0, acc[0][0]);
    acc[0][1] = wmma_f16(a0, b1, acc[0][1]);
    acc[1][0] = wmma_f16(a1, b0, acc[1][0]);
    acc[1][1] = wmma_f16(a1, b1, acc[1][1]);
    __syncthreads();
  }
#endif

#pragma unroll
  for (int r16 = 0; r16 < 2; ++r16)
#pragma unroll
    for (int c16 = 0; c16 < 2; ++c16)
#pragma unroll
      for (int vr = 0; vr < 8; ++vr) {
        int rloc = wm * 32 + r16 * 16 + vr + ((lane >> 4) << 3);
        int nloc = wn * 32 + c16 * 16 + (lane & 15);
        int G = n0g + nloc;
        xproj[((size_t)dir * ROWS_ + r0 + rloc) * G3_ + G] = acc[r16][c16][vr] + bih[G];
      }
}

// ---------------------------------------------------------------------------
// K3: persistent GRU. One 1024-thread workgroup per direction.
// LDS (dynamic, 128KB): h[128][128] f16 + w_hh[384][128] f16.
extern __shared__ char gru_smem[];

__global__ __launch_bounds__(1024) void gru_kernel(
    const _Float16* __restrict__ whh16g, const float* __restrict__ xproj,
    const float* __restrict__ bhhF, const float* __restrict__ bhhB,
    _Float16* __restrict__ h_all) {
  _Float16* h16 = (_Float16*)gru_smem;                  // 128*128 f16
  _Float16* whh = (_Float16*)(gru_smem + H_ * H_ * 2);  // 384*128 f16

  int tid = threadIdx.x, lane = tid & 31, w = tid >> 5;
  int mt = w & 7;        // batch tile
  int jg = w >> 3;       // column group (owns full r/z/n triples for its cols)
  int dir = blockIdx.x;
  const float* xp = xproj + (size_t)dir * ROWS_ * G3_;
  const float* bhh = dir ? bhhB : bhhF;

  const uint32_t* wsrc = (const uint32_t*)(whh16g + (size_t)dir * G3_ * H_);
  uint32_t* wdst = (uint32_t*)whh;
  for (int i = tid; i < G3_ * H_ / 2; i += 1024) wdst[i] = wsrc[i];
  uint32_t* hz = (uint32_t*)h16;
  for (int i = tid; i < H_ * H_ / 2; i += 1024) hz[i] = 0u;
  __syncthreads();

  v8f zz = {};
  for (int t = 0; t < S_; ++t) {
    int tt = dir ? (S_ - 1 - t) : t;

    v8f acc[2][3];
#pragma unroll
    for (int jt = 0; jt < 2; ++jt)
#pragma unroll
      for (int g = 0; g < 3; ++g) acc[jt][g] = zz;

    for (int kc = 0; kc < 4; ++kc) {
      v16h a = frag_a_rm(h16 + (mt * 16) * H_ + kc * 32, H_);
#pragma unroll
      for (int jt = 0; jt < 2; ++jt) {
        int j0 = jg * 32 + jt * 16;
#pragma unroll
        for (int g = 0; g < 3; ++g) {
          v16h bf = frag_b_nt(whh + (g * H_ + j0) * H_ + kc * 32, H_);
          acc[jt][g] = wmma_f16(a, bf, acc[jt][g]);
        }
      }
    }
    __syncthreads();  // all h reads done before h is rewritten

    if (t + 1 < S_) {
      int tn = dir ? (S_ - 2 - t) : (t + 1);
      if (lane == 0)
        __builtin_prefetch(xp + ((size_t)tn * B_ + mt * 16) * G3_, 0, 3);
    }

#pragma unroll
    for (int jt = 0; jt < 2; ++jt) {
      int jj = jg * 32 + jt * 16 + (lane & 15);
      float bhr = bhh[jj], bhz = bhh[H_ + jj], bhn = bhh[2 * H_ + jj];
#pragma unroll
      for (int vr = 0; vr < 8; ++vr) {
        int bb = mt * 16 + vr + ((lane >> 4) << 3);
        const float* xprow = xp + ((size_t)tt * B_ + bb) * G3_;
        float xr = xprow[jj];
        float xzv = xprow[H_ + jj];
        float xn = xprow[2 * H_ + jj];
        float hr = acc[jt][0][vr] + bhr;
        float hzv = acc[jt][1][vr] + bhz;
        float hn = acc[jt][2][vr] + bhn;
        float r = 1.f / (1.f + __expf(-(xr + hr)));
        float z = 1.f / (1.f + __expf(-(xzv + hzv)));
        float n = tanhf(xn + r * hn);
        float hprev = (float)h16[bb * H_ + jj];
        float hnew = (1.f - z) * n + z * hprev;
        h16[bb * H_ + jj] = (_Float16)hnew;
        h_all[((size_t)bb * S_ + tt) * (2 * H_) + dir * H_ + jj] = (_Float16)hnew;
      }
    }
    __syncthreads();  // writes visible before next step's WMMA reads
  }
}

// ---------------------------------------------------------------------------
// K4: u_hat GEMM: [32768 x 256] x [160 x 256]^T -> f32 [B][NC][S][DC]
// Block tile 64x160, 8 waves; K chunks double-buffered via TDM.
__global__ __launch_bounds__(256) void uhat_gemm(
    const _Float16* __restrict__ h_all, const _Float16* __restrict__ wcap_t,
    float* __restrict__ uhat) {
  __shared__ __attribute__((aligned(64))) _Float16 As[2][64 * 32];
  __shared__ __attribute__((aligned(64))) _Float16 Bs[2][160 * 32];

  int tid = threadIdx.x, lane = tid & 31, w = tid >> 5;
  int wm = w & 3, wh = w >> 2;
  size_t r0 = (size_t)blockIdx.x * 64;

  v8f zz = {};
  v8f acc[5];
#pragma unroll
  for (int c = 0; c < 5; ++c) acc[c] = zz;

#if HAVE_TDM
  if (w == 0) {
    tdm_load_2d(As[0], h_all + r0 * 256, 32, 64, 256);
    tdm_load_2d(Bs[0], wcap_t, 32, 160, 256);
    __builtin_amdgcn_s_wait_tensorcnt(0);
  }
  __syncthreads();
  for (int kc = 0; kc < 8; ++kc) {
    int cur = kc & 1, nxt = cur ^ 1;
    if (w == 0 && kc + 1 < 8) {
      tdm_load_2d(As[nxt], h_all + r0 * 256 + (kc + 1) * 32, 32, 64, 256);
      tdm_load_2d(Bs[nxt], wcap_t + (kc + 1) * 32, 32, 160, 256);
    }
    v16h a = frag_a_rm(As[cur] + (wm * 16) * 32, 32);
#pragma unroll
    for (int c = 0; c < 5; ++c) {
      v16h bf = frag_b_nt(Bs[cur] + (wh * 80 + c * 16) * 32, 32);
      acc[c] = wmma_f16(a, bf, acc[c]);
    }
    __syncthreads();
    if (w == 0 && kc + 1 < 8) __builtin_amdgcn_s_wait_tensorcnt(0);
    __syncthreads();
  }
#else
  for (int kc = 0; kc < 8; ++kc) {
    if (tid < 128) {
      int row = tid >> 1, seg = tid & 1;
      *(v16h*)(As[0] + row * 32 + seg * 16) =
          *(const v16h*)(h_all + (r0 + row) * 256 + kc * 32 + seg * 16);
    }
    for (int idx = tid; idx < 320; idx += 256) {
      int row = idx >> 1, seg = idx & 1;
      *(v16h*)(Bs[0] + row * 32 + seg * 16) =
          *(const v16h*)(wcap_t + (size_t)row * 256 + kc * 32 + seg * 16);
    }
    __syncthreads();
    v16h a = frag_a_rm(As[0] + (wm * 16) * 32, 32);
#pragma unroll
    for (int c = 0; c < 5; ++c) {
      v16h bf = frag_b_nt(Bs[0] + (wh * 80 + c * 16) * 32, 32);
      acc[c] = wmma_f16(a, bf, acc[c]);
    }
    __syncthreads();
  }
#endif

#pragma unroll
  for (int c = 0; c < 5; ++c)
#pragma unroll
    for (int vr = 0; vr < 8; ++vr) {
      size_t r = r0 + wm * 16 + vr + ((lane >> 4) << 3);
      int bidx = (int)(r >> 8), s = (int)(r & 255);
      int m = wh * 80 + c * 16 + (lane & 15);
      int nc = m >> 4, dc = m & 15;
      uhat[(((size_t)bidx * NC_ + nc) * S_ + s) * DC_ + dc] = acc[c][vr];
    }
}

// ---------------------------------------------------------------------------
// K5: dynamic routing (5 iters) + final linear. One block per batch element.
__global__ __launch_bounds__(256) void routing_kernel(
    const float* __restrict__ uhat, const float* __restrict__ Wlin,
    const float* __restrict__ blin, float* __restrict__ out) {
  __shared__ float blog[NC_ * S_];
  __shared__ float red[256 * DC_];
  __shared__ float vcap[NC_ * DC_];

  int b = blockIdx.x, tid = threadIdx.x, j = tid;
  const float* uh = uhat + (size_t)b * NC_ * S_ * DC_;

  for (int i = 0; i < NC_; ++i) blog[i * S_ + j] = 0.f;
  __syncthreads();

  for (int it = 0; it < 5; ++it) {
    float c[NC_];
    float mx = blog[j];
    for (int i = 1; i < NC_; ++i) mx = fmaxf(mx, blog[i * S_ + j]);
    float s = 0.f;
    for (int i = 0; i < NC_; ++i) { c[i] = __expf(blog[i * S_ + j] - mx); s += c[i]; }
    float inv = 1.f / s;
    for (int i = 0; i < NC_; ++i) c[i] *= inv;

    for (int i = 0; i < NC_; ++i) {
      const float* up = uh + ((size_t)i * S_ + j) * DC_;
#pragma unroll
      for (int k = 0; k < DC_; ++k) red[tid * DC_ + k] = c[i] * up[k];
      __syncthreads();
      for (int st = 128; st > 0; st >>= 1) {
        if (tid < st) {
#pragma unroll
          for (int k = 0; k < DC_; ++k) red[tid * DC_ + k] += red[(tid + st) * DC_ + k];
        }
        __syncthreads();
      }
      if (tid < DC_) vcap[i * DC_ + tid] = red[tid];
      __syncthreads();
    }

    if (tid < NC_) {
      float s2 = 0.f;
#pragma unroll
      for (int k = 0; k < DC_; ++k) { float v = vcap[tid * DC_ + k]; s2 += v * v; }
      float r = rsqrtf(s2 + 1e-7f);
#pragma unroll
      for (int k = 0; k < DC_; ++k) vcap[tid * DC_ + k] *= r;
    }
    __syncthreads();

    if (it < 4) {
      for (int i = 0; i < NC_; ++i) {
        const float* up = uh + ((size_t)i * S_ + j) * DC_;
        float dot = 0.f;
#pragma unroll
        for (int k = 0; k < DC_; ++k) dot += vcap[i * DC_ + k] * up[k];
        blog[i * S_ + j] += dot;
      }
      __syncthreads();
    }
  }

  if (tid < 2) {
    float acc = blin[tid];
    for (int m = 0; m < NC_ * DC_; ++m) acc += vcap[m] * Wlin[m * 2 + tid];
    out[b * 2 + tid] = acc;
  }
}

// ---------------------------------------------------------------------------
extern "C" void kernel_launch(void* const* d_in, const int* in_sizes, int n_in,
                              void* d_out, int out_size, void* d_ws, size_t ws_size,
                              hipStream_t stream) {
  (void)in_sizes; (void)n_in; (void)out_size; (void)ws_size;
  const int*   x      = (const int*)d_in[0];
  const float* emb    = (const float*)d_in[1];
  const float* w_ih_f = (const float*)d_in[2];
  const float* w_hh_f = (const float*)d_in[3];
  const float* b_ih_f = (const float*)d_in[4];
  const float* b_hh_f = (const float*)d_in[5];
  const float* w_ih_b = (const float*)d_in[6];
  const float* w_hh_b = (const float*)d_in[7];
  const float* b_ih_b = (const float*)d_in[8];
  const float* b_hh_b = (const float*)d_in[9];
  const float* W_cap  = (const float*)d_in[10];
  const float* W_lin  = (const float*)d_in[11];
  const float* b_lin  = (const float*)d_in[12];
  float* out = (float*)d_out;
  char* ws = (char*)d_ws;

  constexpr size_t OFF_EF16   = 0;
  constexpr size_t OFF_WIH16  = OFF_EF16   + (size_t)ROWS_ * DWP_ * 2;
  constexpr size_t OFF_WHH16  = OFF_WIH16  + (size_t)2 * G3_ * DWP_ * 2;
  constexpr size_t OFF_WCAP16 = OFF_WHH16  + (size_t)2 * G3_ * H_ * 2;
  constexpr size_t OFF_XPROJ  = OFF_WCAP16 + (size_t)160 * 256 * 2;
  constexpr size_t OFF_HALL   = OFF_XPROJ  + (size_t)2 * ROWS_ * G3_ * 4;
  constexpr size_t OFF_UHAT   = OFF_HALL   + (size_t)B_ * S_ * 2 * H_ * 2;

  _Float16* ef16    = (_Float16*)(ws + OFF_EF16);
  _Float16* wih16   = (_Float16*)(ws + OFF_WIH16);
  _Float16* whh16   = (_Float16*)(ws + OFF_WHH16);
  _Float16* wcap16t = (_Float16*)(ws + OFF_WCAP16);
  float*    xproj   = (float*)   (ws + OFF_XPROJ);
  _Float16* h_all   = (_Float16*)(ws + OFF_HALL);
  float*    uhat    = (float*)   (ws + OFF_UHAT);

  prep_weights<<<512, 256, 0, stream>>>(w_ih_f, w_ih_b, w_hh_f, w_hh_b, W_cap,
                                        wih16, whh16, wcap16t);
  embed_gather<<<8192, 256, 0, stream>>>(x, emb, ef16);

  dim3 gx(ROWS_ / 128, G3_ / 64, 2);
  xproj_gemm<<<gx, 256, 0, stream>>>(ef16, wih16, b_ih_f, b_ih_b, xproj);

  constexpr int GRU_LDS = (H_ * H_ + G3_ * H_) * 2;  // 128 KB dynamic LDS
  (void)hipFuncSetAttribute((const void*)gru_kernel,
                            hipFuncAttributeMaxDynamicSharedMemorySize, GRU_LDS);
  gru_kernel<<<2, 1024, GRU_LDS, stream>>>(whh16, xproj, b_hh_f, b_hh_b, h_all);

  uhat_gemm<<<ROWS_ / 64, 256, 0, stream>>>(h_all, wcap16t, uhat);

  routing_kernel<<<B_, 256, 0, stream>>>(uhat, W_lin, b_lin, out);
}
